// LSTMModel_5196910428554
// MI455X (gfx1250) — matrix operand order repaired and checked by reading, not compiled, not run on hardware
//
#include <hip/hip_runtime.h>
#include <hip/hip_bf16.h>

typedef __attribute__((ext_vector_type(16))) _Float16 v16h;
typedef __attribute__((ext_vector_type(8)))  float    v8f;

#define LSTM_B 64
#define LSTM_T 512
#define LSTM_I 128
#define LSTM_H 256
#define LSTM_G 1024   // 4*H
#define LSTM_M (LSTM_B * LSTM_T)

// ---------------------------------------------------------------------------
// fragment loaders (per-lane WMMA operand layout, wave32)
// ---------------------------------------------------------------------------
__device__ inline v16h load_frag2(const _Float16* p0, const _Float16* p1) {
  union { v16h v; uint4 q[2]; } u;
  u.q[0] = *reinterpret_cast<const uint4*>(p0);
  u.q[1] = *reinterpret_cast<const uint4*>(p1);
  return u.v;
}

__device__ inline float fast_sigmoid(float x) {
  return 1.0f / (1.0f + __expf(-x));
}
__device__ inline float fast_tanh(float x) {
  float e = __expf(-2.0f * x);
  return (1.0f - e) / (1.0f + e);
}

// ---------------------------------------------------------------------------
// f32 -> f16 elementwise convert
// ---------------------------------------------------------------------------
__global__ void convert_f32_to_f16(const float* __restrict__ src,
                                   _Float16* __restrict__ dst, int n) {
  int i = blockIdx.x * blockDim.x + threadIdx.x;
  if (i < n) dst[i] = (_Float16)src[i];
}

// ---------------------------------------------------------------------------
// Pack W (N x K row-major, f32) into B-fragment order, f16:
//   WB[((kb*N + n)*32) + grp*16 + j] = W[n*K + kb*32 + grp*16 + j]
// so each lane's 16 operand halves are two 16-byte runs at k offsets 8h and 16+8h.
// ---------------------------------------------------------------------------
__global__ void pack_weight(const float* __restrict__ W,
                            _Float16* __restrict__ WB, int K, int N) {
  int idx = blockIdx.x * blockDim.x + threadIdx.x;
  int total = K * N;
  if (idx >= total) return;
  int ii   = idx & 31;          // grp*16 + j
  int rest = idx >> 5;          // kb*N + n
  int n    = rest % N;
  int kb   = rest / N;
  int k    = kb * 32 + ii;
  WB[idx] = (_Float16)W[(size_t)n * K + k];
}

// ---------------------------------------------------------------------------
// Gates GEMM:  G[m][n] = sum_k A[m][k] * Wt[k][n] + b0[n] + b1[n]   (f16 out)
// block = 128 threads (4 waves); tile 64(M) x 64(N); grid = (M/64, N/64)
// ---------------------------------------------------------------------------
__global__ void gemm_gates(const _Float16* __restrict__ A,
                           const _Float16* __restrict__ WB,
                           const float* __restrict__ b0,
                           const float* __restrict__ b1,
                           _Float16* __restrict__ Gout,
                           int K, int N) {
  const int wave = threadIdx.x >> 5;
  const int lane = threadIdx.x & 31;
  const int lr = lane & 15;     // column / row-in-tile selector
  const int lg = lane >> 4;     // lane group
  const int Mbase = blockIdx.x * 64 + wave * 16;
  const int Nbase = blockIdx.y * 64;

  v8f acc[4] = {v8f{0}, v8f{0}, v8f{0}, v8f{0}};

  const _Float16* arow = A + (size_t)(Mbase + lr) * K + lg * 8;
  const int nkb = K >> 5;
  for (int kb = 0; kb < nkb; ++kb) {
    v16h a = load_frag2(arow + kb * 32, arow + kb * 32 + 16);
#pragma unroll
    for (int nt = 0; nt < 4; ++nt) {
      int n = Nbase + nt * 16 + lr;
      const _Float16* bp = WB + ((size_t)kb * N + n) * 32 + lg * 8;
      v16h b = load_frag2(bp, bp + 16);
      acc[nt] = __builtin_amdgcn_wmma_f32_16x16x32_f16(
          false, a, false, b, (short)0, acc[nt], false, false);
    }
  }

#pragma unroll
  for (int nt = 0; nt < 4; ++nt) {
    int col = Nbase + nt * 16 + lr;
    float bias = b0[col] + b1[col];
    int row = Mbase + lg * 8;
#pragma unroll
    for (int r = 0; r < 8; ++r) {
      Gout[(size_t)(row + r) * N + col] = (_Float16)(acc[nt][r] + bias);
    }
  }
}

// ---------------------------------------------------------------------------
// zero h double-buffer + barrier counter
// ---------------------------------------------------------------------------
__global__ void init_state(_Float16* __restrict__ hbuf,
                           unsigned* __restrict__ counter) {
  int i = blockIdx.x * blockDim.x + threadIdx.x;
  if (i < 2 * LSTM_B * LSTM_H) hbuf[i] = (_Float16)0.0f;
  if (i == 0) *counter = 0u;
}

// ---------------------------------------------------------------------------
// Persistent LSTM recurrence.
//   grid = 16 workgroups x 128 threads (4 waves).
//   Workgroup wg owns h columns [wg*16, wg*16+16); its W_hh slice (4 gates x
//   16 cols x K=256) lives in LDS (32 KB) for the full 512 steps.
//   Wave w owns batch rows [16w, 16w+16); its cell state c is a v8f that
//   stays in accumulator registers across all steps.
//   Steps synchronize via atomic-counter device barrier + double-buffered h.
// ---------------------------------------------------------------------------
__global__ void lstm_recurrence(const _Float16* __restrict__ G,  // [B*T][1024]
                                const _Float16* __restrict__ WBhh, // packed
                                _Float16* __restrict__ hbuf,  // [2][B][H]
                                _Float16* __restrict__ hseq,  // [B*T][H] or null
                                float* __restrict__ hlast,    // [B][H]
                                unsigned* __restrict__ counter,
                                int T) {
  __shared__ _Float16 lw[4 * 8 * 16 * 32];   // [gate][kb][n'][32] = 32 KB

  const int tid = threadIdx.x;
  const int wg  = blockIdx.x;
  const int j0  = wg * 16;                   // owned h-column base
  const int wave = tid >> 5;
  const int lane = tid & 31;
  const int lr = lane & 15;
  const int lg = lane >> 4;
  const int Mbase = wave * 16;               // batch-row tile base

  // stage this workgroup's W_hh slice into LDS (B-fragment order)
  for (int i = tid; i < 2048; i += 128) {    // 2048 x 16B chunks
    int ii   = (i & 3) * 8;
    int rest = i >> 2;                       // (gate*8+kb)*16 + n'
    int gate = rest >> 7;
    int kb   = (rest >> 4) & 7;
    int np   = rest & 15;
    int n    = gate * 256 + j0 + np;
    *reinterpret_cast<uint4*>(lw + (size_t)i * 8) =
        *reinterpret_cast<const uint4*>(WBhh + ((size_t)kb * LSTM_G + n) * 32 + ii);
  }
  __syncthreads();

  v8f c = v8f{0};                            // persistent cell state tile

  for (int t = 0; t < T; ++t) {
    const _Float16* hin  = hbuf + (size_t)(t & 1) * LSTM_B * LSTM_H;
    _Float16*       hout = hbuf + (size_t)((t + 1) & 1) * LSTM_B * LSTM_H;

    v8f acc[4] = {v8f{0}, v8f{0}, v8f{0}, v8f{0}};
    const _Float16* arow = hin + (size_t)(Mbase + lr) * LSTM_H + lg * 8;
#pragma unroll
    for (int kb = 0; kb < 8; ++kb) {
      v16h a = load_frag2(arow + kb * 32, arow + kb * 32 + 16);
#pragma unroll
      for (int g = 0; g < 4; ++g) {
        const _Float16* bp = lw + ((size_t)((g * 8 + kb) * 16 + lr)) * 32 + lg * 8;
        v16h b = load_frag2(bp, bp + 16);
        acc[g] = __builtin_amdgcn_wmma_f32_16x16x32_f16(
            false, a, false, b, (short)0, acc[g], false, false);
      }
    }

    // gate math + state update (rows Mbase+lg*8+r, col j0+lr)
    const int col = j0 + lr;
#pragma unroll
    for (int r = 0; r < 8; ++r) {
      int bidx = Mbase + lg * 8 + r;                       // batch index
      const _Float16* gp = G + ((size_t)bidx * T + t) * LSTM_G + col;
      float gi = acc[0][r] + (float)gp[0];
      float gf = acc[1][r] + (float)gp[256];
      float gg = acc[2][r] + (float)gp[512];
      float go = acc[3][r] + (float)gp[768];
      float iv = fast_sigmoid(gi);
      float fv = fast_sigmoid(gf);
      float gv = fast_tanh(gg);
      float ov = fast_sigmoid(go);
      float cv = fv * c[r] + iv * gv;
      c[r] = cv;
      float hv = ov * fast_tanh(cv);
      hout[(size_t)bidx * LSTM_H + col] = (_Float16)hv;
      if (hseq) hseq[((size_t)bidx * T + t) * LSTM_H + col] = (_Float16)hv;
      if (t == T - 1) hlast[(size_t)bidx * LSTM_H + col] = hv;
    }

    // device-wide split barrier (16 co-resident workgroups)
    __threadfence();
    __syncthreads();
    if (tid == 0) {
      __hip_atomic_fetch_add(counter, 1u, __ATOMIC_RELEASE,
                             __HIP_MEMORY_SCOPE_AGENT);
      unsigned target = 16u * (unsigned)(t + 1);
      while (__hip_atomic_load(counter, __ATOMIC_ACQUIRE,
                               __HIP_MEMORY_SCOPE_AGENT) < target) {
        __builtin_amdgcn_s_sleep(1);
      }
    }
    __syncthreads();
  }
}

// ---------------------------------------------------------------------------
// final FC (256 -> 3) + softmax, 64 threads (one per batch row)
// ---------------------------------------------------------------------------
__global__ void fc_softmax(const float* __restrict__ h,
                           const float* __restrict__ fcw,
                           const float* __restrict__ fcb,
                           float* __restrict__ out) {
  int b = threadIdx.x;
  if (b >= LSTM_B) return;
  float z[3];
#pragma unroll
  for (int cc = 0; cc < 3; ++cc) {
    float s = fcb[cc];
    for (int k = 0; k < LSTM_H; ++k)
      s += fcw[cc * LSTM_H + k] * h[(size_t)b * LSTM_H + k];
    z[cc] = s;
  }
  float m = fmaxf(z[0], fmaxf(z[1], z[2]));
  float e0 = __expf(z[0] - m), e1 = __expf(z[1] - m), e2 = __expf(z[2] - m);
  float inv = 1.0f / (e0 + e1 + e2);
  out[b * 3 + 0] = e0 * inv;
  out[b * 3 + 1] = e1 * inv;
  out[b * 3 + 2] = e2 * inv;
}

// ---------------------------------------------------------------------------
extern "C" void kernel_launch(void* const* d_in, const int* in_sizes, int n_in,
                              void* d_out, int out_size, void* d_ws, size_t ws_size,
                              hipStream_t stream) {
  const float* x      = (const float*)d_in[0];
  const float* W_ih_0 = (const float*)d_in[1];
  const float* W_hh_0 = (const float*)d_in[2];
  const float* b_ih_0 = (const float*)d_in[3];
  const float* b_hh_0 = (const float*)d_in[4];
  const float* W_ih_1 = (const float*)d_in[5];
  const float* W_hh_1 = (const float*)d_in[6];
  const float* b_ih_1 = (const float*)d_in[7];
  const float* b_hh_1 = (const float*)d_in[8];
  const float* fc_w   = (const float*)d_in[9];
  const float* fc_b   = (const float*)d_in[10];
  float* out = (float*)d_out;

  // workspace carve-up
  char* ws = (char*)d_ws;
  size_t off = 0;
  auto take = [&](size_t bytes) {
    char* p = ws + off;
    off += (bytes + 255) & ~(size_t)255;
    return p;
  };
  _Float16* x_h   = (_Float16*)take((size_t)LSTM_M * LSTM_I * 2);   // 8 MB
  _Float16* hseq  = (_Float16*)take((size_t)LSTM_M * LSTM_H * 2);   // 16 MB
  _Float16* G     = (_Float16*)take((size_t)LSTM_M * LSTM_G * 2);   // 64 MB
  _Float16* WBih0 = (_Float16*)take((size_t)LSTM_I * LSTM_G * 2);
  _Float16* WBhh0 = (_Float16*)take((size_t)LSTM_H * LSTM_G * 2);
  _Float16* WBih1 = (_Float16*)take((size_t)LSTM_H * LSTM_G * 2);
  _Float16* WBhh1 = (_Float16*)take((size_t)LSTM_H * LSTM_G * 2);
  _Float16* hbuf  = (_Float16*)take((size_t)2 * LSTM_B * LSTM_H * 2);
  float*    hlast = (float*)take((size_t)LSTM_B * LSTM_H * 4);
  unsigned* counter = (unsigned*)take(256);
  (void)ws_size; (void)in_sizes; (void)n_in; (void)out_size;

  // 1. convert / pack
  convert_f32_to_f16<<<(LSTM_M * LSTM_I + 255) / 256, 256, 0, stream>>>(
      x, x_h, LSTM_M * LSTM_I);
  pack_weight<<<(LSTM_I * LSTM_G + 255) / 256, 256, 0, stream>>>(
      W_ih_0, WBih0, LSTM_I, LSTM_G);
  pack_weight<<<(LSTM_H * LSTM_G + 255) / 256, 256, 0, stream>>>(
      W_hh_0, WBhh0, LSTM_H, LSTM_G);
  pack_weight<<<(LSTM_H * LSTM_G + 255) / 256, 256, 0, stream>>>(
      W_ih_1, WBih1, LSTM_H, LSTM_G);
  pack_weight<<<(LSTM_H * LSTM_G + 255) / 256, 256, 0, stream>>>(
      W_hh_1, WBhh1, LSTM_H, LSTM_G);

  dim3 ggrid(LSTM_M / 64, LSTM_G / 64);

  // 2. layer 0: input-projection GEMM, then persistent recurrence
  gemm_gates<<<ggrid, 128, 0, stream>>>(x_h, WBih0, b_ih_0, b_hh_0, G,
                                        LSTM_I, LSTM_G);
  init_state<<<128, 256, 0, stream>>>(hbuf, counter);
  lstm_recurrence<<<16, 128, 0, stream>>>(G, WBhh0, hbuf, hseq, hlast,
                                          counter, LSTM_T);

  // 3. layer 1 (gates buffer reused)
  gemm_gates<<<ggrid, 128, 0, stream>>>(hseq, WBih1, b_ih_1, b_hh_1, G,
                                        LSTM_H, LSTM_G);
  init_state<<<128, 256, 0, stream>>>(hbuf, counter);
  lstm_recurrence<<<16, 128, 0, stream>>>(G, WBhh1, hbuf, (_Float16*)nullptr,
                                          hlast, counter, LSTM_T);

  // 4. classifier head
  fc_softmax<<<1, 64, 0, stream>>>(hlast, fc_w, fc_b, out);
}
